// LocalAttentionBlock_18245021073730
// MI455X (gfx1250) — matrix operand order repaired and checked by reading, not compiled, not run on hardware
//
#include <hip/hip_runtime.h>

// ---------------------------------------------------------------------------
// Complex local-attention block for MI455X (gfx1250), wave32 + WMMA bf16.
// GEMM-shaped work (QKV/O projections, QK^T, PV) -> v_wmma_f32_16x16x32_bf16.
// LDS staging uses GLOBAL_LOAD_ASYNC_TO_LDS_B128 + s_wait_asynccnt with
// double buffering when the toolchain exposes the builtins.
// ---------------------------------------------------------------------------

#define DEVI __device__ __forceinline__

typedef __attribute__((ext_vector_type(16))) __bf16 v16bf;
typedef __attribute__((ext_vector_type(8)))  float  v8f;
typedef int v4i __attribute__((vector_size(16)));

union FragA { unsigned int u[8]; v16bf v; };

static constexpr int BATCH = 2;
static constexpr int SEQ   = 1024;
static constexpr int HID   = 1024;
static constexpr int NH    = 16;
static constexpr int HD    = 64;
static constexpr int WIN   = 4096;
static constexpr int CACHE = 3072;

static constexpr int M1 = BATCH * SEQ;   // 2048 rows
static constexpr int KK = 2 * HID;       // 2048 (re|im concat)
static constexpr int NN = HID;           // 1024 cols

#if defined(__HIP_DEVICE_COMPILE__) && \
    __has_builtin(__builtin_amdgcn_global_load_async_to_lds_b128) && \
    __has_builtin(__builtin_amdgcn_s_wait_asynccnt)
#define USE_ASYNC_LDS 1
#else
#define USE_ASYNC_LDS 0
#endif

#define AS1 __attribute__((address_space(1)))
#define AS3 __attribute__((address_space(3)))

DEVI void async_b128(const unsigned int* g, unsigned int* l) {
#if USE_ASYNC_LDS
  __builtin_amdgcn_global_load_async_to_lds_b128(
      (AS1 v4i*)(unsigned long long)(const void*)g,
      (AS3 v4i*)(unsigned long long)(void*)l, 0, 0);
#else
  (void)g; (void)l;
#endif
}
DEVI void wait_async() {
#if USE_ASYNC_LDS
  __builtin_amdgcn_s_wait_asynccnt(0);
#endif
}

DEVI unsigned short f2bf(float f) {
  unsigned int u = __float_as_uint(f);
  u += 0x7fffu + ((u >> 16) & 1u);       // round-to-nearest-even
  return (unsigned short)(u >> 16);
}
DEVI unsigned int pack2(unsigned short lo, unsigned short hi) {
  return (unsigned int)lo | ((unsigned int)hi << 16);
}

// ---------------------------------------------------------------------------
// 1) Complex LayerNorm -> packed bf16 A1 [2048][2048]  (cols 0..1023 = re)
// ---------------------------------------------------------------------------
__global__ __launch_bounds__(256) void ln_kernel(
    const float* __restrict__ hre, const float* __restrict__ him,
    const float* __restrict__ gamma, const float* __restrict__ bre,
    const float* __restrict__ bim, unsigned short* __restrict__ A1) {
  __shared__ float sbuf[256];
  const int m   = blockIdx.x;
  const int tid = threadIdx.x;
  const float* xr = hre + (size_t)m * HID;
  const float* xi = him + (size_t)m * HID;

  float sr = 0.f, si = 0.f;
  for (int k = tid; k < HID; k += 256) { sr += xr[k]; si += xi[k]; }
  sbuf[tid] = sr; __syncthreads();
  for (int s = 128; s > 0; s >>= 1) { if (tid < s) sbuf[tid] += sbuf[tid + s]; __syncthreads(); }
  const float mur = sbuf[0] * (1.0f / HID); __syncthreads();
  sbuf[tid] = si; __syncthreads();
  for (int s = 128; s > 0; s >>= 1) { if (tid < s) sbuf[tid] += sbuf[tid + s]; __syncthreads(); }
  const float mui = sbuf[0] * (1.0f / HID); __syncthreads();

  float ss = 0.f;
  for (int k = tid; k < HID; k += 256) {
    float cr = xr[k] - mur, ci = xi[k] - mui;
    ss += cr * cr + ci * ci;
  }
  sbuf[tid] = ss; __syncthreads();
  for (int s = 128; s > 0; s >>= 1) { if (tid < s) sbuf[tid] += sbuf[tid + s]; __syncthreads(); }
  const float inv = rsqrtf(sbuf[0] * (1.0f / HID) + 1e-5f);

  unsigned short* row = A1 + (size_t)m * KK;
  for (int k = tid; k < HID; k += 256) {
    float cr = xr[k] - mur, ci = xi[k] - mui;
    float g = gamma[k];
    row[k]       = f2bf(cr * inv * g + bre[k]);
    row[HID + k] = f2bf(ci * inv * g + bim[k]);
  }
}

// ---------------------------------------------------------------------------
// 2) Pack complex weight stacks into bf16-pair u32 [K/2][N]:
//    B_r = [Wr ; -Wi],  B_i = [Wi ; Wr]   (for y = x @ W.T with x=[re|im])
// ---------------------------------------------------------------------------
__global__ __launch_bounds__(256) void pack_w_kernel(
    const float* __restrict__ Wr, const float* __restrict__ Wi,
    unsigned int* __restrict__ Bpr, unsigned int* __restrict__ Bpi) {
  const int total = (KK / 2) * NN; // 1024*1024
  for (int idx = blockIdx.x * blockDim.x + threadIdx.x; idx < total;
       idx += gridDim.x * blockDim.x) {
    int n  = idx >> 10;
    int kk = idx & 1023;
    float a0, a1, c0, c1;
    if (kk < 512) {           // k in [0,1024): re part of x
      int k0 = 2 * kk;
      a0 =  Wr[(size_t)n * HID + k0];  a1 =  Wr[(size_t)n * HID + k0 + 1];
      c0 =  Wi[(size_t)n * HID + k0];  c1 =  Wi[(size_t)n * HID + k0 + 1];
    } else {                  // k in [1024,2048): im part of x
      int k0 = 2 * kk - HID;
      a0 = -Wi[(size_t)n * HID + k0];  a1 = -Wi[(size_t)n * HID + k0 + 1];
      c0 =  Wr[(size_t)n * HID + k0];  c1 =  Wr[(size_t)n * HID + k0 + 1];
    }
    Bpr[(size_t)kk * NN + n] = pack2(f2bf(a0), f2bf(a1));
    Bpi[(size_t)kk * NN + n] = pack2(f2bf(c0), f2bf(c1));
  }
}

// ---------------------------------------------------------------------------
// 3) Copy KV caches into d_out rows [0,3072) and produce bf16 copies
// ---------------------------------------------------------------------------
__global__ __launch_bounds__(256) void cache_copy_kernel(
    const float* __restrict__ Kcr, const float* __restrict__ Kci,
    const float* __restrict__ Vcr, const float* __restrict__ Vci,
    float* __restrict__ oKr, float* __restrict__ oKi,
    float* __restrict__ oVr, float* __restrict__ oVi,
    unsigned short* __restrict__ bKr, unsigned short* __restrict__ bKi,
    unsigned short* __restrict__ bVr, unsigned short* __restrict__ bVi) {
  const int total = BATCH * CACHE * HID;
  for (int idx = blockIdx.x * blockDim.x + threadIdx.x; idx < total;
       idx += gridDim.x * blockDim.x) {
    int b   = idx / (CACHE * HID);
    int rem = idx - b * (CACHE * HID);
    int t   = rem >> 10;
    int n   = rem & 1023;
    size_t dst = ((size_t)b * WIN + t) * HID + n;
    float v;
    v = Kcr[idx]; oKr[dst] = v; bKr[dst] = f2bf(v);
    v = Kci[idx]; oKi[dst] = v; bKi[dst] = f2bf(v);
    v = Vcr[idx]; oVr[dst] = v; bVr[dst] = f2bf(v);
    v = Vci[idx]; oVi[dst] = v; bVi[dst] = f2bf(v);
  }
}

// ---------------------------------------------------------------------------
// 4) WMMA GEMM:  C[M][N] = A_bf16[M][K] * B_bf16[K][N] + bias[N]
//    A as u32 K-pairs [M][K/2]; B as u32 K-pairs [K/2][N].
//    Block 128x128, BK=32, 8 waves (4 along M x 2 along N), 32x64 per wave.
//    LDS: As[buf][m][kk] stride 20, Bs[buf][kk][n] stride 132 (both b128-
//    aligned rows). Async double-buffered when available.
//    mode 0: bf16 out [M][N]      (Q projections)
//    mode 1: f32 + bf16 out at rows b*4096+3072+s  (K/V projections)
//    mode 2: f32 out = addend + acc                 (O projection + residual)
// ---------------------------------------------------------------------------
__global__ __launch_bounds__(256) void gemm_bf16_kernel(
    const unsigned int* __restrict__ A32, const unsigned int* __restrict__ B32,
    const float* __restrict__ bias, int M, int N, int K, int mode,
    float* __restrict__ out_f, unsigned short* __restrict__ out_bf,
    const float* __restrict__ addend) {
  __shared__ unsigned int As[2][128][20];    // rows 80B apart (16B aligned)
  __shared__ unsigned int Bs[2][16][132];    // rows 528B apart (16B aligned)

  const int tid  = threadIdx.x;
  const int lane = tid & 31;
  const int wave = tid >> 5;
  const int wm = wave & 3;
  const int wn = wave >> 2;
  const int m0 = blockIdx.y * 128;
  const int n0 = blockIdx.x * 128;
  const int K2 = K >> 1;
  const int l16   = lane & 15;
  const int half8 = (lane < 16) ? 0 : 8;
  const int aoff1 = (lane < 16) ? 0 : 8;
  const int aoff2 = (lane < 16) ? 4 : 12;
  const int boff  = (lane < 16) ? 0 : 8;

  v8f acc[2][4];
  for (int i = 0; i < 2; ++i)
    for (int j = 0; j < 4; ++j)
      for (int r = 0; r < 8; ++r) acc[i][j][r] = 0.f;

#if USE_ASYNC_LDS
  // prologue: async-stage tile 0 into buffer 0
  {
    for (int i = tid; i < 512; i += 256) {                 // A: 128 rows x 4 xb128
      int r = i >> 2, c4 = (i & 3) << 2;
      async_b128(A32 + (size_t)(m0 + r) * K2 + c4, &As[0][r][c4]);
    }
    for (int i = tid; i < 512; i += 256) {                 // B: 16 rows x 32 xb128
      int r = i >> 5, c4 = (i & 31) << 2;
      async_b128(B32 + (size_t)r * N + n0 + c4, &Bs[0][r][c4]);
    }
  }
  int cur = 0;
#endif

  for (int kt = 0; kt < K; kt += 32) {
#if USE_ASYNC_LDS
    wait_async();
    __syncthreads();
    if (kt + 32 < K) {
      const int kk1 = (kt + 32) >> 1;
      const int nxt = cur ^ 1;
      for (int i = tid; i < 512; i += 256) {
        int r = i >> 2, c4 = (i & 3) << 2;
        async_b128(A32 + (size_t)(m0 + r) * K2 + kk1 + c4, &As[nxt][r][c4]);
      }
      for (int i = tid; i < 512; i += 256) {
        int r = i >> 5, c4 = (i & 31) << 2;
        async_b128(B32 + (size_t)(kk1 + r) * N + n0 + c4, &Bs[nxt][r][c4]);
      }
    }
    const unsigned int (*Asc)[20]  = As[cur];
    const unsigned int (*Bsc)[132] = Bs[cur];
    cur ^= 1;
#else
    const int kk0 = kt >> 1;
    __syncthreads();
    for (int i = tid; i < 128 * 16; i += 256) {            // A tile
      int r = i >> 4, c = i & 15;
      As[0][r][c] = A32[(size_t)(m0 + r) * K2 + kk0 + c];
    }
    for (int i = tid; i < 16 * 128; i += 256) {            // B tile [kk][n]
      int r = i >> 7, n = i & 127;
      Bs[0][r][n] = B32[(size_t)(kk0 + r) * N + n0 + n];
    }
    __syncthreads();
    const unsigned int (*Asc)[20]  = As[0];
    const unsigned int (*Bsc)[132] = Bs[0];
#endif

#pragma unroll
    for (int mi = 0; mi < 2; ++mi) {
      FragA a;
      const int row = wm * 32 + mi * 16 + l16;
#pragma unroll
      for (int v = 0; v < 4; ++v) a.u[v] = Asc[row][aoff1 + v];
#pragma unroll
      for (int v = 0; v < 4; ++v) a.u[4 + v] = Asc[row][aoff2 + v];
#pragma unroll
      for (int ni = 0; ni < 4; ++ni) {
        FragA bf;
        const int col = wn * 64 + ni * 16 + l16;
#pragma unroll
        for (int v = 0; v < 8; ++v) bf.u[v] = Bsc[boff + v][col];
        acc[mi][ni] = __builtin_amdgcn_wmma_f32_16x16x32_bf16(
            false, a.v, false, bf.v, (short)0, acc[mi][ni], false, false);
      }
    }
  }

#pragma unroll
  for (int mi = 0; mi < 2; ++mi) {
#pragma unroll
    for (int ni = 0; ni < 4; ++ni) {
      const int gn = n0 + wn * 64 + ni * 16 + l16;
      const float bv = bias[gn];
#pragma unroll
      for (int r = 0; r < 8; ++r) {
        const int gm = m0 + wm * 32 + mi * 16 + r + half8;
        const float val = acc[mi][ni][r] + bv;
        if (mode == 0) {
          out_bf[(size_t)gm * N + gn] = f2bf(val);
        } else if (mode == 1) {
          const int b = gm >> 10, s = gm & 1023;
          const size_t idx = ((size_t)(b * WIN + CACHE + s)) * N + gn;
          out_f[idx]  = val;
          out_bf[idx] = f2bf(val);
        } else {
          const size_t idx = (size_t)gm * N + gn;
          out_f[idx] = addend[idx] + val;
        }
      }
    }
  }
}

// ---------------------------------------------------------------------------
// 5) Flash attention: one block = (b, h, 64-query tile), 4 waves x 16 queries.
//    scores = (Qr Kr^T + Qi Ki^T)/8, mask t <= q+3072, online softmax,
//    O_c += P V_c. Output -> packed bf16 A2 [2048][2048] for O projection.
// ---------------------------------------------------------------------------
__global__ __launch_bounds__(128) void attn_kernel(
    const unsigned int* __restrict__ Qr32, const unsigned int* __restrict__ Qi32,
    const unsigned short* __restrict__ Krb, const unsigned short* __restrict__ Kib,
    const unsigned short* __restrict__ Vrb, const unsigned short* __restrict__ Vib,
    unsigned short* __restrict__ A2) {
  __shared__ unsigned int Qs[2][64][36];   // [c][q][hd-pair]   18.0 KB
  __shared__ unsigned int Ks[2][32][36];   // [c][t][hd-pair]    9.0 KB
  __shared__ unsigned int Vs[2][64][20];   // [c][hd][key-pair] 10.0 KB
  __shared__ unsigned short Ps[64][40];    // [q][key]           5.0 KB

  const int tid  = threadIdx.x;
  const int lane = tid & 31;
  const int wq   = tid >> 5;
  const int q0   = blockIdx.x * 64;
  const int h    = blockIdx.y;
  const int b    = blockIdx.z;
  const int l16   = lane & 15;
  const int half8 = (lane < 16) ? 0 : 8;
  const int aoff1 = (lane < 16) ? 0 : 8;
  const int aoff2 = (lane < 16) ? 4 : 12;
  const int boff  = (lane < 16) ? 0 : 8;

  const unsigned int* Kr32 = (const unsigned int*)Krb;
  const unsigned int* Ki32 = (const unsigned int*)Kib;

  // --- load Q tile (both components) ---
#if USE_ASYNC_LDS
  for (int i = tid; i < 2 * 64 * 8; i += 128) {            // 1024 b128 moves
    int c = i >> 9, rem = i & 511;
    int row = rem >> 3, c4 = (rem & 7) << 2;
    const unsigned int* src = c ? Qi32 : Qr32;
    async_b128(src + (size_t)(b * SEQ + q0 + row) * (HID / 2) + h * 32 + c4,
               &Qs[c][row][c4]);
  }
  wait_async();
#else
  for (int i = tid; i < 2 * 64 * 32; i += 128) {
    int c = i >> 11, rem = i & 2047;
    int row = rem >> 5, kk = rem & 31;
    const unsigned int* src = c ? Qi32 : Qr32;
    Qs[c][row][kk] = src[(size_t)(b * SEQ + q0 + row) * (HID / 2) + h * 32 + kk];
  }
#endif
  __syncthreads();

  FragA qa[2][2];
  const int arow = wq * 16 + l16;
#pragma unroll
  for (int c = 0; c < 2; ++c)
#pragma unroll
    for (int w = 0; w < 2; ++w) {
#pragma unroll
      for (int v = 0; v < 4; ++v) qa[c][w].u[v] = Qs[c][arow][w * 16 + aoff1 + v];
#pragma unroll
      for (int v = 0; v < 4; ++v) qa[c][w].u[4 + v] = Qs[c][arow][w * 16 + aoff2 + v];
    }

  v8f Oacc[2][4];
  for (int c = 0; c < 2; ++c)
    for (int nc = 0; nc < 4; ++nc)
      for (int r = 0; r < 8; ++r) Oacc[c][nc][r] = 0.f;
  float m_st[8], l_st[8];
  for (int r = 0; r < 8; ++r) { m_st[r] = -1e30f; l_st[r] = 0.f; }

  const int nkb = (q0 + 3136) >> 5;        // #32-key blocks (mask handles edges)
  for (int kb = 0; kb < nkb; ++kb) {
    const int t0 = kb * 32;
    __syncthreads();
    // --- stage K tiles (async when available; V pack overlaps the DMA) ---
#if USE_ASYNC_LDS
    for (int i = tid; i < 2 * 32 * 8; i += 128) {          // 512 b128 moves
      int c = i >> 8, rem = i & 255;
      int tt = rem >> 3, c4 = (rem & 7) << 2;
      const unsigned int* src = c ? Ki32 : Kr32;
      async_b128(src + (size_t)(b * WIN + t0 + tt) * (HID / 2) + h * 32 + c4,
                 &Ks[c][tt][c4]);
    }
#else
    for (int i = tid; i < 2 * 32 * 32; i += 128) {
      int c = i >> 10, rem = i & 1023;
      int tt = rem >> 5, kk = rem & 31;
      const unsigned int* src = c ? Ki32 : Kr32;
      Ks[c][tt][kk] = src[(size_t)(b * WIN + t0 + tt) * (HID / 2) + h * 32 + kk];
    }
#endif
    for (int i = tid; i < 2 * 64 * 16; i += 128) {         // V key-pair pack
      int c = i >> 10, rem = i & 1023;
      int n = rem >> 4, kk = rem & 15;
      const unsigned short* src = c ? Vib : Vrb;
      size_t base = (size_t)(b * WIN + t0 + 2 * kk) * HID + h * HD + n;
      Vs[c][n][kk] = pack2(src[base], src[base + HID]);
    }
    wait_async();
    __syncthreads();

    // --- scores: two 16-key tiles ---
    v8f sc[2];
#pragma unroll
    for (int tt2 = 0; tt2 < 2; ++tt2) {
      for (int r = 0; r < 8; ++r) sc[tt2][r] = 0.f;
      const int trow = tt2 * 16 + l16;
#pragma unroll
      for (int c = 0; c < 2; ++c)
#pragma unroll
        for (int w = 0; w < 2; ++w) {
          FragA kf;
#pragma unroll
          for (int v = 0; v < 8; ++v) kf.u[v] = Ks[c][trow][w * 16 + boff + v];
          sc[tt2] = __builtin_amdgcn_wmma_f32_16x16x32_bf16(
              false, qa[c][w].v, false, kf.v, (short)0, sc[tt2], false, false);
        }
#pragma unroll
      for (int r = 0; r < 8; ++r) {
        const int q = q0 + wq * 16 + r + half8;
        const int t = t0 + tt2 * 16 + l16;
        float sv = sc[tt2][r] * 0.125f;
        if (t > q + CACHE) sv = -1e30f;
        sc[tt2][r] = sv;
      }
    }

    // --- online softmax (row stats replicated across each 16-lane half) ---
#pragma unroll
    for (int r = 0; r < 8; ++r) {
      float mx = fmaxf(sc[0][r], sc[1][r]);
      for (int off = 1; off < 16; off <<= 1) mx = fmaxf(mx, __shfl_xor(mx, off, 32));
      const float mn = fmaxf(m_st[r], mx);
      const float corr = __expf(m_st[r] - mn);
      const float p0 = __expf(sc[0][r] - mn);
      const float p1 = __expf(sc[1][r] - mn);
      float rs = p0 + p1;
      for (int off = 1; off < 16; off <<= 1) rs += __shfl_xor(rs, off, 32);
      l_st[r] = l_st[r] * corr + rs;
      m_st[r] = mn;
#pragma unroll
      for (int c = 0; c < 2; ++c)
#pragma unroll
        for (int nc = 0; nc < 4; ++nc) Oacc[c][nc][r] *= corr;
      const int prow = wq * 16 + r + half8;
      Ps[prow][l16]      = f2bf(p0);
      Ps[prow][16 + l16] = f2bf(p1);
    }
    __syncthreads();

    // --- O += P * V ---
    const unsigned int* Ps32 = (const unsigned int*)&Ps[0][0];
    FragA pa;
    const int pbase = (wq * 16 + l16) * 20;                // row stride 40 shorts
#pragma unroll
    for (int v = 0; v < 4; ++v) pa.u[v] = Ps32[pbase + aoff1 + v];
#pragma unroll
    for (int v = 0; v < 4; ++v) pa.u[4 + v] = Ps32[pbase + aoff2 + v];
#pragma unroll
    for (int c = 0; c < 2; ++c)
#pragma unroll
      for (int nc = 0; nc < 4; ++nc) {
        FragA vf;
        const int vn = nc * 16 + l16;
#pragma unroll
        for (int v = 0; v < 8; ++v) vf.u[v] = Vs[c][vn][boff + v];
        Oacc[c][nc] = __builtin_amdgcn_wmma_f32_16x16x32_bf16(
            false, pa.v, false, vf.v, (short)0, Oacc[c][nc], false, false);
      }
  }

  // --- normalize + write packed bf16 A2 [2048][2048] (re | im cols) ---
#pragma unroll
  for (int r = 0; r < 8; ++r) {
    const float inv = 1.0f / l_st[r];
    const int gm = b * SEQ + q0 + wq * 16 + r + half8;
#pragma unroll
    for (int c = 0; c < 2; ++c)
#pragma unroll
      for (int nc = 0; nc < 4; ++nc) {
        const int col = c * HID + h * HD + nc * 16 + l16;
        A2[(size_t)gm * KK + col] = f2bf(Oacc[c][nc][r] * inv);
      }
  }
}

// ---------------------------------------------------------------------------
// Host-side orchestration
// ---------------------------------------------------------------------------
extern "C" void kernel_launch(void* const* d_in, const int* in_sizes, int n_in,
                              void* d_out, int out_size, void* d_ws, size_t ws_size,
                              hipStream_t stream) {
  (void)in_sizes; (void)n_in; (void)out_size; (void)ws_size;

  const float* hre = (const float*)d_in[0];
  const float* him = (const float*)d_in[1];
  const float* Kcr = (const float*)d_in[2];
  const float* Kci = (const float*)d_in[3];
  const float* Vcr = (const float*)d_in[4];
  const float* Vci = (const float*)d_in[5];
  const float* gamma = (const float*)d_in[6];
  const float* bre   = (const float*)d_in[7];
  const float* bim   = (const float*)d_in[8];
  const float* qWr = (const float*)d_in[9];  const float* qWi = (const float*)d_in[10];
  const float* qbr = (const float*)d_in[11]; const float* qbi = (const float*)d_in[12];
  const float* kWr = (const float*)d_in[13]; const float* kWi = (const float*)d_in[14];
  const float* kbr = (const float*)d_in[15]; const float* kbi = (const float*)d_in[16];
  const float* vWr = (const float*)d_in[17]; const float* vWi = (const float*)d_in[18];
  const float* vbr = (const float*)d_in[19]; const float* vbi = (const float*)d_in[20];
  const float* oWr = (const float*)d_in[21]; const float* oWi = (const float*)d_in[22];
  const float* obr = (const float*)d_in[23]; const float* obi = (const float*)d_in[24];

  // d_out regions (f32), flat in return order
  float* out_re = (float*)d_out;                         // 2*1024*1024
  float* out_im = out_re + (size_t)BATCH * SEQ * HID;
  float* nKr = out_im + (size_t)BATCH * SEQ * HID;       // 2*4096*1024 each
  float* nKi = nKr + (size_t)BATCH * WIN * HID;
  float* nVr = nKi + (size_t)BATCH * WIN * HID;
  float* nVi = nVr + (size_t)BATCH * WIN * HID;

  // workspace layout
  char* w = (char*)d_ws;
  unsigned short* A1 = (unsigned short*)w;                       w += (size_t)M1 * KK * 2;   // 8 MB
  unsigned int* Bp[8];
  for (int i = 0; i < 8; ++i) { Bp[i] = (unsigned int*)w;        w += (size_t)(KK/2) * NN * 4; } // 32 MB
  unsigned short* Qrb = (unsigned short*)w;                      w += (size_t)M1 * NN * 2;   // 4 MB
  unsigned short* Qib = (unsigned short*)w;                      w += (size_t)M1 * NN * 2;
  unsigned short* Krb = (unsigned short*)w;                      w += (size_t)BATCH * WIN * HID * 2; // 16 MB
  unsigned short* Kib = (unsigned short*)w;                      w += (size_t)BATCH * WIN * HID * 2;
  unsigned short* Vrb = (unsigned short*)w;                      w += (size_t)BATCH * WIN * HID * 2;
  unsigned short* Vib = (unsigned short*)w;                      w += (size_t)BATCH * WIN * HID * 2;
  unsigned short* A2  = (unsigned short*)w;                      w += (size_t)M1 * KK * 2;   // 8 MB

  // 1) LayerNorm -> A1
  ln_kernel<<<M1, 256, 0, stream>>>(hre, him, gamma, bre, bim, A1);

  // 2) Weight packing (q,k,v,o)
  pack_w_kernel<<<2048, 256, 0, stream>>>(qWr, qWi, Bp[0], Bp[1]);
  pack_w_kernel<<<2048, 256, 0, stream>>>(kWr, kWi, Bp[2], Bp[3]);
  pack_w_kernel<<<2048, 256, 0, stream>>>(vWr, vWi, Bp[4], Bp[5]);
  pack_w_kernel<<<2048, 256, 0, stream>>>(oWr, oWi, Bp[6], Bp[7]);

  // 3) Cache copy into outputs + bf16 staging
  cache_copy_kernel<<<8192, 256, 0, stream>>>(Kcr, Kci, Vcr, Vci,
                                              nKr, nKi, nVr, nVi,
                                              Krb, Kib, Vrb, Vib);

  // 4) Projections (M=2048, N=1024, K=2048)
  const unsigned int* A1u = (const unsigned int*)A1;
  dim3 gg(NN / 128, M1 / 128);
  gemm_bf16_kernel<<<gg, 256, 0, stream>>>(A1u, Bp[0], qbr, M1, NN, KK, 0, nullptr, Qrb, nullptr);
  gemm_bf16_kernel<<<gg, 256, 0, stream>>>(A1u, Bp[1], qbi, M1, NN, KK, 0, nullptr, Qib, nullptr);
  gemm_bf16_kernel<<<gg, 256, 0, stream>>>(A1u, Bp[2], kbr, M1, NN, KK, 1, nKr, Krb, nullptr);
  gemm_bf16_kernel<<<gg, 256, 0, stream>>>(A1u, Bp[3], kbi, M1, NN, KK, 1, nKi, Kib, nullptr);
  gemm_bf16_kernel<<<gg, 256, 0, stream>>>(A1u, Bp[4], vbr, M1, NN, KK, 1, nVr, Vrb, nullptr);
  gemm_bf16_kernel<<<gg, 256, 0, stream>>>(A1u, Bp[5], vbi, M1, NN, KK, 1, nVi, Vib, nullptr);

  // 5) Attention -> A2
  attn_kernel<<<dim3(SEQ / 64, NH, BATCH), 128, 0, stream>>>(
      (const unsigned int*)Qrb, (const unsigned int*)Qib, Krb, Kib, Vrb, Vib, A2);

  // 6) Output projection + residual
  const unsigned int* A2u = (const unsigned int*)A2;
  gemm_bf16_kernel<<<gg, 256, 0, stream>>>(A2u, Bp[6], obr, M1, NN, KK, 2, out_re, nullptr, hre);
  gemm_bf16_kernel<<<gg, 256, 0, stream>>>(A2u, Bp[7], obi, M1, NN, KK, 2, out_im, nullptr, him);
}
